// MaskWindowAttention_43258910605747
// MI455X (gfx1250) — matrix operand order repaired and checked by reading, not compiled, not run on hardware
//
#include <hip/hip_runtime.h>

// ---------------------------------------------------------------------------
// MaskWindowAttention for MI455X (gfx1250): bf16 WMMA (16x16x32) everywhere,
// fp32 accumulate; async global->LDS staging where the copy is a pure
// byte-move (q/k tiles, proj input tile). Shapes hardcoded from setup_inputs().
// ---------------------------------------------------------------------------

typedef __bf16 bf16_t;
typedef __attribute__((ext_vector_type(16))) __bf16 v16bf;
typedef __attribute__((ext_vector_type(8)))  __bf16 v8bf;
typedef __attribute__((ext_vector_type(8)))  float  v8f;
typedef __attribute__((ext_vector_type(4)))  int    v4i;

// Problem constants
constexpr int Bb  = 2;     // batch
constexpr int T   = 3;     // frames
constexpr int HH  = 192;   // height
constexpr int WWp = 192;   // width
constexpr int C   = 180;   // dim
constexpr int NH  = 6;     // heads
constexpr int HD  = 30;    // head dim
constexpr int NW_H = 24, NW_W = 24;          // windows per spatial dim
constexpr int NWIN = Bb * NW_H * NW_W;       // 1152 windows
constexpr int N   = 192;   // tokens per window (3*8*8)
constexpr int NQ  = 64;    // query tokens (last frame)
constexpr int CK  = 192;   // channel dim padded to mult. of 32 (K for WMMA)
constexpr int QKV = 540;   // 3*C
constexpr int QKVP = 544;  // padded to mult. of 16

// ---- float -> bf16 (round to nearest even) --------------------------------
__device__ __forceinline__ unsigned short f2bf(float f) {
    union { float f; unsigned u; } c; c.f = f;
    unsigned u = c.u;
    return (unsigned short)((u + 0x7FFFu + ((u >> 16) & 1u)) >> 16);
}

// ---- async global->LDS copy (16B granules) --------------------------------
// Uses GLOBAL_LOAD_ASYNC_TO_LDS_B128 (ASYNCcnt) when the toolchain exposes
// the builtin; falls back to a plain generic copy otherwise. The builtin's
// params are typed: (addrspace(1) v4i* src, addrspace(3) v4i* dst, imm, imm).
__device__ __forceinline__ void async_copy_b128(const void* g, void* lds) {
#if defined(__has_builtin) && __has_builtin(__builtin_amdgcn_global_load_async_to_lds_b128)
    __builtin_amdgcn_global_load_async_to_lds_b128(
        (__attribute__((address_space(1))) v4i*)(g),
        (__attribute__((address_space(3))) v4i*)(lds),
        /*imm offset=*/0, /*cpol=*/0);
#else
    *(v4i*)lds = *(const v4i*)g;
#endif
}

__device__ __forceinline__ void wait_async_copies() {
#if defined(__has_builtin) && __has_builtin(__builtin_amdgcn_s_wait_asynccnt)
    __builtin_amdgcn_s_wait_asynccnt(0);
#else
    asm volatile("s_wait_asynccnt 0" ::: "memory");
#endif
}

// ---- WMMA helpers ---------------------------------------------------------
__device__ __forceinline__ v8f wmma_bf16(v16bf a, v16bf b, v8f c) {
    // D = A(16x32 bf16) x B(32x16 bf16) + C(16x16 f32)
    return __builtin_amdgcn_wmma_f32_16x16x32_bf16(
        /*neg_a=*/false, a, /*neg_b=*/false, b,
        /*c_mod=*/(short)0, c, /*reuse_a=*/false, /*reuse_b=*/false);
}

// A fragment (16x32 bf16), row-major source. ISA layout: lane half hl holds
// K = hl*8 + j (j<8) and K = 16 + hl*8 + (j-8) (j>=8) for row M = lane%16.
__device__ __forceinline__ v16bf load_a_frag(const unsigned short* base,
                                             int rowStride, int row, int k0,
                                             int lane) {
    int hl = (lane >> 4) & 1;
    const unsigned short* p = base + (size_t)row * rowStride + k0 + hl * 8;
    union { v16bf v; v8bf h[2]; } u;
    u.h[0] = *(const v8bf*)(p);
    u.h[1] = *(const v8bf*)(p + 16);
    return u.v;
}

// B fragment (32x16 bf16): lane covers column N = lane%16; holds K = k0 +
// hl*8 + j (j<8) and K = k0 + 16 + hl*8 + (j-8) (j>=8) from operand row `nrow`
// (operand stored so that column n's K values are contiguous = row n).
__device__ __forceinline__ v16bf load_b_frag(const unsigned short* base,
                                             int rowStride, int nrow, int k0,
                                             int lane) {
    int hl = (lane >> 4) & 1;
    const unsigned short* p = base + (size_t)nrow * rowStride + k0 + hl * 8;
    union { v16bf v; v8bf h[2]; } u;
    u.h[0] = *(const v8bf*)(p);
    u.h[1] = *(const v8bf*)(p + 16);
    return u.v;
}

// ---------------------------------------------------------------------------
// Kernel 0: convert + zero-pad weights to bf16.
//   wqkv_p : [QKVP=544][CK=192] bf16 (rows>=540 and k>=180 zero)
//   wproj_p: [CK=192][CK=192]  bf16 (rows>=180 and k>=180 zero)
// ---------------------------------------------------------------------------
__global__ void conv_weights_kernel(const float* __restrict__ w_qkv,
                                    const float* __restrict__ w_proj,
                                    unsigned short* __restrict__ wqkv_p,
                                    unsigned short* __restrict__ wproj_p) {
    int i = blockIdx.x * blockDim.x + threadIdx.x;
    if (i < QKVP * CK) {
        int n = i / CK, k = i % CK;
        float v = (n < QKV && k < C) ? w_qkv[n * C + k] : 0.f;
        wqkv_p[i] = f2bf(v);
    }
    if (i < CK * CK) {
        int n = i / CK, k = i % CK;
        float v = (n < C && k < C) ? w_proj[n * C + k] : 0.f;
        wproj_p[i] = f2bf(v);
    }
}

// ---------------------------------------------------------------------------
// Kernel 1: per-window gather (with roll) + QKV GEMM.
//   writes q_ws [win][h][64][32] (scaled by hd^-0.5), k_ws/v_ws [win][h][192][32]
//   (bf16, d=30,31 zero) and pred_q (fp32, raw window tokens 128..191).
// ---------------------------------------------------------------------------
__global__ __launch_bounds__(256) void qkv_kernel(
    const float* __restrict__ x, const float* __restrict__ b_qkv,
    const unsigned short* __restrict__ wqkv_p,
    const long long* __restrict__ shift,
    unsigned short* __restrict__ q_ws, unsigned short* __restrict__ k_ws,
    unsigned short* __restrict__ v_ws, float* __restrict__ pred_q_out) {
    __shared__ __align__(32) unsigned short xs[N][CK]; // 73,728 B

    int win = blockIdx.x;
    int b   = win / (NW_H * NW_W);
    int wh  = (win / NW_W) % NW_H;
    int ww  = win % NW_W;
    int s0 = (int)shift[0], s1 = (int)shift[1], s2 = (int)shift[2];

    // Gather rolled window tokens -> bf16 LDS; emit pred_q for tokens >= 128.
    for (int idx = threadIdx.x; idx < N * C; idx += blockDim.x) {
        int n = idx / C, c = idx % C;
        int nd = n >> 6, nh = (n >> 3) & 7, nw = n & 7;
        int ts = (nd + s0) % T;
        int hs = (wh * 8 + nh + s1) % HH;
        int wsrc = (ww * 8 + nw + s2) % WWp;
        float v = x[(((size_t)(b * T + ts) * HH + hs) * WWp + wsrc) * C + c];
        xs[n][c] = f2bf(v);
        if (n >= 2 * NQ)
            pred_q_out[((size_t)win * NQ + (n - 2 * NQ)) * C + c] = v;
    }
    for (int idx = threadIdx.x; idx < N * (CK - C); idx += blockDim.x) {
        xs[idx / (CK - C)][C + idx % (CK - C)] = 0;  // channel pad
    }
    // Zero hd-pad (d=30,31) of q/k/v scratch for this window.
    for (int idx = threadIdx.x; idx < NH * NQ * 2; idx += blockDim.x) {
        int h = idx / (NQ * 2); int r = (idx >> 1) % NQ; int d = 30 + (idx & 1);
        q_ws[(((size_t)win * NH + h) * NQ + r) * 32 + d] = 0;
    }
    for (int idx = threadIdx.x; idx < NH * N * 2; idx += blockDim.x) {
        int h = idx / (N * 2); int r = (idx >> 1) % N; int d = 30 + (idx & 1);
        k_ws[(((size_t)win * NH + h) * N + r) * 32 + d] = 0;
        v_ws[(((size_t)win * NH + h) * N + r) * 32 + d] = 0;
    }
    __syncthreads();

    int wave = threadIdx.x >> 5, lane = threadIdx.x & 31;
    int hl = lane >> 4, ln16 = lane & 15;
    const float scale = 0.18257418583505536f; // 30^-0.5

    // Output tiles: 12 (M=192 tokens) x 34 (N=544 channels), K = 6 x 32.
    for (int tile = wave; tile < 12 * 34; tile += 8) {
        int tm = tile % 12, tn = tile / 12;
        v8f acc = {};
        int arow = tm * 16 + ln16;
        int brow = tn * 16 + ln16;
        for (int kb = 0; kb < 6; ++kb) {
            v16bf a = load_a_frag(&xs[0][0], CK, arow, kb * 32, lane);
            v16bf bf = load_b_frag(wqkv_p, CK, brow, kb * 32, lane);
            acc = wmma_bf16(a, bf, acc);
        }
        int ncol = tn * 16 + ln16;
        if (ncol < QKV) {
            float bias = b_qkv[ncol];
            int part = ncol / C;          // 0=q 1=k 2=v
            int ch   = ncol % C;
            int head = ch / HD;
            int d    = ch % HD;
#pragma unroll
            for (int r = 0; r < 8; ++r) {
                int m = tm * 16 + r + 8 * hl;   // token index
                float val = acc[r] + bias;
                if (part == 0) {
                    if (m >= 2 * NQ)
                        q_ws[(((size_t)win * NH + head) * NQ + (m - 2 * NQ)) * 32 + d]
                            = f2bf(val * scale);
                } else if (part == 1) {
                    k_ws[(((size_t)win * NH + head) * N + m) * 32 + d] = f2bf(val);
                } else {
                    v_ws[(((size_t)win * NH + head) * N + m) * 32 + d] = f2bf(val);
                }
            }
        }
    }
}

// ---------------------------------------------------------------------------
// Kernel 2: per-(window, head) attention.
//   q/k tiles staged with async global->LDS (pure byte-move, ASYNCcnt);
//   V is transposed on the way into LDS (manual), overlapping the async copy.
//   S = q kT (single WMMA per tile: K=32 covers hd) + analytic rel-pos bias,
//   fp32 softmax, P (bf16) @ V, write o_ws [win][64][192].
// ---------------------------------------------------------------------------
__global__ __launch_bounds__(256) void attn_kernel(
    const unsigned short* __restrict__ q_ws,
    const unsigned short* __restrict__ k_ws,
    const unsigned short* __restrict__ v_ws,
    const float* __restrict__ rpb,
    unsigned short* __restrict__ o_ws) {
    __shared__ __align__(32) unsigned short qs[NQ][32];   //  4 KB
    __shared__ __align__(32) unsigned short ks[N][32];    // 12 KB
    __shared__ __align__(32) unsigned short vst[32][N];   // 12 KB (transposed)
    __shared__ __align__(32) float          Sf[NQ][N];    // 48 KB
    __shared__ __align__(32) unsigned short ps[NQ][N];    // 24 KB

    int win  = blockIdx.x / NH;
    int head = blockIdx.x % NH;
    const unsigned short* qg = q_ws + ((size_t)win * NH + head) * NQ * 32;
    const unsigned short* kg = k_ws + ((size_t)win * NH + head) * N * 32;
    const unsigned short* vg = v_ws + ((size_t)win * NH + head) * N * 32;

    // q: 256 x 16B granules, one per thread; k: 768 granules, 3 per thread.
    {
        int t = threadIdx.x;
        async_copy_b128((const char*)qg + t * 16, (char*)&qs[0][0] + t * 16);
#pragma unroll
        for (int j = 0; j < 3; ++j) {
            int cidx = t + j * 256;
            async_copy_b128((const char*)kg + cidx * 16,
                            (char*)&ks[0][0] + cidx * 16);
        }
    }
    // V transpose (needs per-element shuffle -> manual path, overlaps asyncs).
    for (int i = threadIdx.x; i < N * 32; i += blockDim.x) {
        vst[i & 31][i >> 5] = vg[i];   // [d][token]
    }
    wait_async_copies();
    __syncthreads();

    int wave = threadIdx.x >> 5, lane = threadIdx.x & 31;
    int hl = lane >> 4, ln16 = lane & 15;

    // S tiles: 4 (M=64) x 12 (N=192), one WMMA each (K=32 >= hd).
    for (int tile = wave; tile < 4 * 12; tile += 8) {
        int tm = tile % 4, tn = tile / 4;
        v8f acc = {};
        v16bf a  = load_a_frag(&qs[0][0], 32, tm * 16 + ln16, 0, lane);
        v16bf bf = load_b_frag(&ks[0][0], 32, tn * 16 + ln16, 0, lane);
        acc = wmma_bf16(a, bf, acc);
        int n  = tn * 16 + ln16;                 // key token
        int kd = n >> 6, kh = (n >> 3) & 7, kw = n & 7;
#pragma unroll
        for (int r = 0; r < 8; ++r) {
            int m  = tm * 16 + r + 8 * hl;       // query token (0..63)
            int qh = m >> 3, qw = m & 7;
            // REL_IDX closed form: rd = 2-kd, rh = qh-kh+7, rw = qw-kw+7
            int idxb = (((2 - kd) * 225) + (qh - kh + 7) * 15 + (qw - kw + 7)) * NH + head;
            Sf[m][n] = acc[r] + rpb[idxb];
        }
    }
    __syncthreads();

    // Softmax over keys, one row per thread (threads 0..63).
    if (threadIdx.x < NQ) {
        int m = threadIdx.x;
        float mx = -1e30f;
        for (int n = 0; n < N; ++n) mx = fmaxf(mx, Sf[m][n]);
        float sum = 0.f;
        for (int n = 0; n < N; ++n) {
            float e = __expf(Sf[m][n] - mx);
            Sf[m][n] = e;
            sum += e;
        }
        float inv = 1.f / sum;
        for (int n = 0; n < N; ++n) ps[m][n] = f2bf(Sf[m][n] * inv);
    }
    __syncthreads();

    // O = P (64x192) @ V (192x32): 4 x 2 tiles, one tile per wave, K = 6 x 32.
    {
        int tm = wave % 4, tn = wave / 4;
        v8f acc = {};
        for (int kb = 0; kb < 6; ++kb) {
            v16bf a  = load_a_frag(&ps[0][0], N, tm * 16 + ln16, kb * 32, lane);
            v16bf bf = load_b_frag(&vst[0][0], N, tn * 16 + ln16, kb * 32, lane);
            acc = wmma_bf16(a, bf, acc);
        }
        int d = tn * 16 + ln16;
        if (d < HD) {
#pragma unroll
            for (int r = 0; r < 8; ++r) {
                int m = tm * 16 + r + 8 * hl;
                o_ws[((size_t)win * NQ + m) * CK + head * HD + d] = f2bf(acc[r]);
            }
        }
    }
    // head 5 also zeroes channel pad 180..191 of this window's o_ws rows.
    if (head == NH - 1) {
        for (int i = threadIdx.x; i < NQ * (CK - C); i += blockDim.x) {
            int m = i / (CK - C), c = C + i % (CK - C);
            o_ws[((size_t)win * NQ + m) * CK + c] = 0;
        }
    }
}

// ---------------------------------------------------------------------------
// Kernel 3: proj GEMM + fused roll-back scatter into xo (181 ch) + x_pred.
//   Input tile staged via async global->LDS (24 KB linear byte-move).
// ---------------------------------------------------------------------------
__global__ __launch_bounds__(256) void proj_kernel(
    const unsigned short* __restrict__ o_ws,
    const unsigned short* __restrict__ wproj_p,
    const float* __restrict__ b_proj,
    const long long* __restrict__ shift,
    float* __restrict__ xo, float* __restrict__ x_pred) {
    __shared__ __align__(32) unsigned short xs2[NQ][CK]; // 24,576 B

    int win = blockIdx.x;
    {
        const char* src = (const char*)(o_ws + (size_t)win * NQ * CK);
        char* dst = (char*)&xs2[0][0];
#pragma unroll
        for (int j = 0; j < 6; ++j) {             // 1536 granules / 256 thr
            int cidx = threadIdx.x + j * 256;
            async_copy_b128(src + cidx * 16, dst + cidx * 16);
        }
    }
    wait_async_copies();
    __syncthreads();

    int s1 = (int)shift[1], s2 = (int)shift[2];
    int b  = win / (NW_H * NW_W);
    int wh = (win / NW_W) % NW_H;
    int ww = win % NW_W;
    int wave = threadIdx.x >> 5, lane = threadIdx.x & 31;
    int hl = lane >> 4, ln16 = lane & 15;

    // Tiles: 4 (M=64) x 12 (N=192), K = 6 x 32.
    for (int tile = wave; tile < 4 * 12; tile += 8) {
        int tm = tile % 4, tn = tile / 4;
        v8f acc = {};
        for (int kb = 0; kb < 6; ++kb) {
            v16bf a  = load_a_frag(&xs2[0][0], CK, tm * 16 + ln16, kb * 32, lane);
            v16bf bf = load_b_frag(wproj_p, CK, tn * 16 + ln16, kb * 32, lane);
            acc = wmma_bf16(a, bf, acc);
        }
        int n = tn * 16 + ln16;
        float bias = (n < C) ? b_proj[n] : 0.f;
#pragma unroll
        for (int r = 0; r < 8; ++r) {
            int m = tm * 16 + r + 8 * hl;         // query token 0..63
            float val = acc[r] + bias;
            int mh = m >> 3, mw = m & 7;
            int ho = (wh * 8 + mh + s1) % HH;
            int wo = (ww * 8 + mw + s2) % WWp;
            size_t obase = (((size_t)b * HH + ho) * WWp + wo) * (size_t)(C + 1);
            if (n < C) {
                xo[obase + n] = val;
                x_pred[((size_t)win * NQ + m) * C + n] = val;
            } else if (n == C) {
                xo[obase + C] = 1.0f;             // pred-flag column
            }
        }
    }
}

// ---------------------------------------------------------------------------
extern "C" void kernel_launch(void* const* d_in, const int* in_sizes, int n_in,
                              void* d_out, int out_size, void* d_ws, size_t ws_size,
                              hipStream_t stream) {
    (void)in_sizes; (void)n_in; (void)out_size; (void)ws_size;
    const float* x      = (const float*)d_in[0];
    const float* w_qkv  = (const float*)d_in[1];
    const float* b_qkv  = (const float*)d_in[2];
    const float* w_proj = (const float*)d_in[3];
    const float* b_proj = (const float*)d_in[4];
    const float* rpb    = (const float*)d_in[5];
    const long long* shift = (const long long*)d_in[6];

    // d_out layout: xo (2*192*192*181) | pred_q (1152*64*180) | x_pred (same)
    float* xo     = (float*)d_out;
    float* pred_q = xo + (size_t)Bb * HH * WWp * (C + 1);
    float* x_pred = pred_q + (size_t)NWIN * NQ * C;

    // Workspace carve (bf16 scratch, ~227 MB total)
    char* ws = (char*)d_ws;
    auto alignup = [](size_t v) { return (v + 255) & ~(size_t)255; };
    size_t off = 0;
    unsigned short* wqkv_p  = (unsigned short*)(ws + off); off = alignup(off + (size_t)QKVP * CK * 2);
    unsigned short* wproj_p = (unsigned short*)(ws + off); off = alignup(off + (size_t)CK * CK * 2);
    unsigned short* q_ws    = (unsigned short*)(ws + off); off = alignup(off + (size_t)NWIN * NH * NQ * 32 * 2);
    unsigned short* k_ws    = (unsigned short*)(ws + off); off = alignup(off + (size_t)NWIN * NH * N * 32 * 2);
    unsigned short* v_ws    = (unsigned short*)(ws + off); off = alignup(off + (size_t)NWIN * NH * N * 32 * 2);
    unsigned short* o_ws    = (unsigned short*)(ws + off); off = alignup(off + (size_t)NWIN * NQ * CK * 2);

    conv_weights_kernel<<<(QKVP * CK + 255) / 256, 256, 0, stream>>>(
        w_qkv, w_proj, wqkv_p, wproj_p);
    qkv_kernel<<<NWIN, 256, 0, stream>>>(
        x, b_qkv, wqkv_p, shift, q_ws, k_ws, v_ws, pred_q);
    attn_kernel<<<NWIN * NH, 256, 0, stream>>>(
        q_ws, k_ws, v_ws, rpb, o_ws);
    proj_kernel<<<NWIN, 256, 0, stream>>>(
        o_ws, wproj_p, b_proj, shift, xo, x_pred);
}
